// LSTM_53893249630610
// MI455X (gfx1250) — matrix operand order repaired and checked by reading, not compiled, not run on hardware
//
#include <hip/hip_runtime.h>

// ---------------------------------------------------------------------------
// LSTM (F=64, H=3, B=2048, T=512) for gfx1250.
// Phase 1: zx[B*T, 16] = x[B*T, 64] @ W[64, 12 (pad 16)] via v_wmma_f32_16x16x32_f16
// Phase 2: per-batch-element sequential recurrence + fused dense head.
// ---------------------------------------------------------------------------

typedef __attribute__((ext_vector_type(16))) _Float16 v16h;
typedef __attribute__((ext_vector_type(8)))  float    v8f;

#define F_IN    64
#define HID     3
#define G4      12            // 4*H
#define BATCH   2048
#define TSTEPS  512
#define NROWS   (BATCH * TSTEPS)   // 1,048,576
#define NTILES  (NROWS / 16)       // 65,536
#define ZSTRIDE 16                 // padded z row (12 useful + 4 zero)

// ---------------------------------------------------------------------------
// Phase 1: batched input projection with WMMA.
// One wave computes 16x16 output tiles (16 rows of x, 16 padded gate cols).
// K = 64 -> two 16x16x32 f16 WMMA ops per tile, accumulated in f32.
// ---------------------------------------------------------------------------
__global__ void __launch_bounds__(256)
lstm_gemm_xW(const float* __restrict__ x,
             const float* __restrict__ W,
             float* __restrict__ zx)
{
    const int lane          = threadIdx.x & 31;
    const int waveInBlock   = threadIdx.x >> 5;
    const int wavesPerBlock = blockDim.x >> 5;
    const int waveId        = blockIdx.x * wavesPerBlock + waveInBlock;
    const int numWaves      = gridDim.x * wavesPerBlock;

    const int n  = lane & 15;   // output column (gate index; >=12 is padding)
    const int hi = lane >> 4;   // lane half-group

    // B fragments (32x16 f16): lane holds column n, halfs h = K - 16*hi (+32*c).
    // Columns >= 12 are zero so the padded accumulator columns stay zero.
    v16h b0, b1;
#pragma unroll
    for (int h = 0; h < 16; ++h) {
        const int k0 = 16 * hi + h;        // chunk 0: K in [0,32)
        const int k1 = 32 + 16 * hi + h;   // chunk 1: K in [32,64)
        const float w0 = (n < G4) ? W[k0 * G4 + n] : 0.0f;
        const float w1 = (n < G4) ? W[k1 * G4 + n] : 0.0f;
        b0[h] = (_Float16)w0;
        b1[h] = (_Float16)w1;
    }

    for (int tile = waveId; tile < NTILES; tile += numWaves) {
        const int m = lane & 15;           // row within tile
        const float* row = x + (size_t)(tile * 16 + m) * F_IN;

        // A fragments (16x32 f16): half h -> K = 16*hi + h (+32*c).
        // Two contiguous 8-float runs per chunk -> vectorizable loads + cvt.
        v16h a0, a1;
#pragma unroll
        for (int h = 0; h < 8; ++h) {
            a0[h]     = (_Float16)row[     16 * hi + h];
            a0[h + 8] = (_Float16)row[ 8 + 16 * hi + h];
            a1[h]     = (_Float16)row[32 + 16 * hi + h];
            a1[h + 8] = (_Float16)row[40 + 16 * hi + h];
        }

        v8f acc = {};
        acc = __builtin_amdgcn_wmma_f32_16x16x32_f16(
                  false, a0, false, b0, (short)0, acc, false, false);
        acc = __builtin_amdgcn_wmma_f32_16x16x32_f16(
                  false, a1, false, b1, (short)0, acc, false, false);

        // D layout: VGPR r, lane L -> D[M = r + 8*(L/16)][N = L%16]
        float* orow = zx + (size_t)tile * 16 * ZSTRIDE;
#pragma unroll
        for (int r = 0; r < 8; ++r) {
            const int mm = r + 8 * hi;
            orow[(size_t)mm * ZSTRIDE + n] = acc[r];
        }
    }
}

// ---------------------------------------------------------------------------
// Phase 2: sequential recurrence, one lane per batch element.
// ---------------------------------------------------------------------------
__device__ __forceinline__ float fast_sigmoid(float v) {
    return 1.0f / (1.0f + __expf(-v));
}
__device__ __forceinline__ float fast_tanh(float v) {
    // 1 - 2/(e^{2v}+1): saturates cleanly to +/-1 at large |v|.
    const float e = __expf(2.0f * v);
    return 1.0f - 2.0f / (e + 1.0f);
}

__global__ void __launch_bounds__(256)
lstm_recurrence(const float* __restrict__ zx,
                const float* __restrict__ U,     // [3, 12]
                const float* __restrict__ bias,  // [12]
                const float* __restrict__ Wd,    // [3]
                const float* __restrict__ bd,    // [1]
                float* __restrict__ out)         // [2048]
{
    const int b = blockIdx.x * blockDim.x + threadIdx.x;
    if (b >= BATCH) return;

    // Tiny uniform constants -> registers (scalar loads; L2 resident).
    float u[HID][G4];
#pragma unroll
    for (int k = 0; k < HID; ++k)
#pragma unroll
        for (int j = 0; j < G4; ++j)
            u[k][j] = U[k * G4 + j];

    float bz[G4];
#pragma unroll
    for (int j = 0; j < G4; ++j) bz[j] = bias[j];

    float h0 = 0.f, h1 = 0.f, h2 = 0.f;
    float c0 = 0.f, c1 = 0.f, c2 = 0.f;

    const float* zrow = zx + (size_t)b * TSTEPS * ZSTRIDE;

    for (int t = 0; t < TSTEPS; ++t) {
        const float4 q0 = *(const float4*)(zrow);
        const float4 q1 = *(const float4*)(zrow + 4);
        const float4 q2 = *(const float4*)(zrow + 8);
        float z[G4] = { q0.x, q0.y, q0.z, q0.w,
                        q1.x, q1.y, q1.z, q1.w,
                        q2.x, q2.y, q2.z, q2.w };
#pragma unroll
        for (int j = 0; j < G4; ++j) {
            z[j] += bz[j];
            z[j] = fmaf(h0, u[0][j], z[j]);
            z[j] = fmaf(h1, u[1][j], z[j]);
            z[j] = fmaf(h2, u[2][j], z[j]);
        }
        // Keras gate order: i, f, g, o
        const float i0 = fast_sigmoid(z[0]), i1 = fast_sigmoid(z[1]), i2 = fast_sigmoid(z[2]);
        const float f0 = fast_sigmoid(z[3]), f1 = fast_sigmoid(z[4]), f2 = fast_sigmoid(z[5]);
        const float g0 = fast_tanh(z[6]),    g1 = fast_tanh(z[7]),    g2 = fast_tanh(z[8]);
        const float o0 = fast_sigmoid(z[9]), o1 = fast_sigmoid(z[10]), o2 = fast_sigmoid(z[11]);

        c0 = fmaf(f0, c0, i0 * g0);
        c1 = fmaf(f1, c1, i1 * g1);
        c2 = fmaf(f2, c2, i2 * g2);
        h0 = o0 * fast_tanh(c0);
        h1 = o1 * fast_tanh(c1);
        h2 = o2 * fast_tanh(c2);

        zrow += ZSTRIDE;
    }

    float y = bd[0];
    y = fmaf(h0, Wd[0], y);
    y = fmaf(h1, Wd[1], y);
    y = fmaf(h2, Wd[2], y);
    out[b] = fast_sigmoid(y);
}

// ---------------------------------------------------------------------------
// Launch
// ---------------------------------------------------------------------------
extern "C" void kernel_launch(void* const* d_in, const int* in_sizes, int n_in,
                              void* d_out, int out_size, void* d_ws, size_t ws_size,
                              hipStream_t stream) {
    const float* x  = (const float*)d_in[0];   // [B, T, F]
    const float* W  = (const float*)d_in[1];   // [F, 4H]
    const float* U  = (const float*)d_in[2];   // [H, 4H]
    const float* b  = (const float*)d_in[3];   // [4H]
    const float* Wd = (const float*)d_in[4];   // [H, 1]
    const float* bd = (const float*)d_in[5];   // [1]
    float* out = (float*)d_out;                // [B, 1] f32
    float* zx  = (float*)d_ws;                 // needs B*T*16*4 = 64 MB

    // Phase 1: 65536 tiles over 16384 waves (grid-stride), 8 waves / block.
    lstm_gemm_xW<<<2048, 256, 0, stream>>>(x, W, zx);

    // Phase 2: one lane per batch element.
    lstm_recurrence<<<(BATCH + 255) / 256, 256, 0, stream>>>(zx, U, b, Wd, bd, out);
}
